// Attention_46875273068626
// MI455X (gfx1250) — matrix operand order repaired and checked by reading, not compiled, not run on hardware
//
#include <hip/hip_runtime.h>
#include <hip/hip_bf16.h>

typedef _Float16 v16h  __attribute__((ext_vector_type(16)));
typedef _Float16 half8 __attribute__((ext_vector_type(8)));
typedef _Float16 half4 __attribute__((ext_vector_type(4)));
typedef float    v8f   __attribute__((ext_vector_type(8)));

#define S_LEN 2048
#define DDIM  512
#define QT    32      // Q rows per block
#define KT    64      // keys per tile
#define NWARP 8

// LDS byte layout
//  Qs   : QT*DDIM*2      = 32768
//  Ks   : KT*DDIM*2      = 65536
//  Vt   : DDIM*KT*2      = 65536
//  Sraw : QT*KT*4        = 8192
//  Ps   : QT*KT*2        = 4096
//  m,l,sc: 3*QT*4        = 384
#define SMEM_BYTES (32768 + 65536 + 65536 + 8192 + 4096 + 384)

// ---- WMMA fragment loaders (layouts per CDNA5 ISA 7.12.2, wave32) ----
// A (16x32 f16): lane m=lane&15; low lanes hold K=[0..7]+[16..23], high lanes K=[8..15]+[24..31]
__device__ __forceinline__ v16h load_frag_a(const _Float16* p, int ld, int rowBase, int kBase, int lane) {
  int m  = lane & 15;
  int hi = lane >> 4;
  const _Float16* q = p + (rowBase + m) * ld + kBase + (hi << 3);
  half8 lo = *(const half8*)(q);
  half8 hv = *(const half8*)(q + 16);
  return __builtin_shufflevector(lo, hv, 0,1,2,3,4,5,6,7,8,9,10,11,12,13,14,15);
}
// B (32x16 f16): lane n=lane&15 holds one column; low lanes K=[0..7]+[16..23], high lanes K=[8..15]+[24..31]
__device__ __forceinline__ v16h load_frag_b(const _Float16* p, int ld, int colBase, int kBase, int lane) {
  int n  = lane & 15;
  int hi = lane >> 4;
  const _Float16* q = p + (colBase + n) * ld + kBase + (hi << 3);
  half8 lo = *(const half8*)(q);
  half8 hv = *(const half8*)(q + 16);
  return __builtin_shufflevector(lo, hv, 0,1,2,3,4,5,6,7,8,9,10,11,12,13,14,15);
}

__device__ __forceinline__ v8f wmma_f16(v16h a, v16h b, v8f c) {
  return __builtin_amdgcn_wmma_f32_16x16x32_f16(false, a, false, b, (short)0, c, false, false);
}

__global__ __launch_bounds__(256, 1)
void attn_flash_kernel(const float* __restrict__ x, float* __restrict__ out) {
  extern __shared__ char smem[];
  _Float16* Qs   = (_Float16*)smem;                     // [QT][DDIM]
  _Float16* Ks   = Qs + QT * DDIM;                      // [KT][DDIM] row-major
  _Float16* Vt   = Ks + KT * DDIM;                      // [DDIM][KT] transposed
  float*    Sraw = (float*)(Vt + DDIM * KT);            // [QT][KT]
  _Float16* Ps   = (_Float16*)(Sraw + QT * KT);         // [QT][KT]
  float*    mArr = (float*)(Ps + QT * KT);              // [QT]
  float*    lArr = mArr + QT;                           // [QT]
  float*    scArr= lArr + QT;                           // [QT]

  const int tid  = threadIdx.x;
  const int lane = tid & 31;
  const int wid  = tid >> 5;
  const int b    = blockIdx.y;
  const int q0   = blockIdx.x * QT;
  const float* xb = x + (size_t)b * S_LEN * DDIM;

  // ---- Load + convert Q tile to LDS (f16) ----
  #pragma unroll
  for (int i = 0; i < 16; ++i) {
    int idx = i * 256 + tid;           // 4096 float4 total
    int row = idx >> 7;                // 128 float4 per row
    int c4  = idx & 127;
    int d   = c4 * 4;
    float4 f = *(const float4*)(xb + (size_t)(q0 + row) * DDIM + d);
    half4 h;
    h.x = (_Float16)f.x; h.y = (_Float16)f.y; h.z = (_Float16)f.z; h.w = (_Float16)f.w;
    *(half4*)(Qs + row * DDIM + d) = h;
  }
  if (tid < QT) { mArr[tid] = -1e30f; lArr[tid] = 0.0f; }

  // ---- Output accumulators: 2 M-tiles x 4 N-tiles of 16x16 per wave ----
  v8f accO[8];
  #pragma unroll
  for (int t = 0; t < 8; ++t)
    accO[t] = (v8f){0.f,0.f,0.f,0.f,0.f,0.f,0.f,0.f};

  const int mt = wid >> 2;       // score-stage M tile (0..1)
  const int ct = wid & 3;        // score-stage N tile (0..3)
  const int dbase = wid * 64;    // PV-stage D slice

  #pragma unroll 1
  for (int kt = 0; kt < S_LEN / KT; ++kt) {
    const int kbase = kt * KT;
    __syncthreads();   // previous iteration done reading Ks/Vt/Ps

    // ---- Load + convert K tile: row-major Ks and transposed Vt ----
    #pragma unroll 4
    for (int i = 0; i < 32; ++i) {
      int idx = i * 256 + tid;         // 8192 float4 total
      int row = idx >> 7;
      int c4  = idx & 127;
      int d   = c4 * 4;
      float4 f = *(const float4*)(xb + (size_t)(kbase + row) * DDIM + d);
      half4 h;
      h.x = (_Float16)f.x; h.y = (_Float16)f.y; h.z = (_Float16)f.z; h.w = (_Float16)f.w;
      *(half4*)(Ks + row * DDIM + d) = h;
      Vt[(d + 0) * KT + row] = h.x;
      Vt[(d + 1) * KT + row] = h.y;
      Vt[(d + 2) * KT + row] = h.z;
      Vt[(d + 3) * KT + row] = h.w;
    }
    // prefetch next K tile into L2
    if (kt + 1 < S_LEN / KT) {
      #pragma unroll
      for (int p = 0; p < 4; ++p)
        __builtin_prefetch(xb + (size_t)(kbase + KT) * DDIM + (p * 256 + tid) * 32, 0, 0);
    }
    __syncthreads();

    // ---- Scores: S = Q @ K^T  (one 16x16 tile per wave, k over 512) ----
    v8f s0 = (v8f){0.f,0.f,0.f,0.f,0.f,0.f,0.f,0.f};
    v8f s1 = (v8f){0.f,0.f,0.f,0.f,0.f,0.f,0.f,0.f};
    #pragma unroll
    for (int ks = 0; ks < 16; ks += 2) {
      v16h a0 = load_frag_a(Qs, DDIM, mt * 16, ks * 32, lane);
      v16h b0 = load_frag_b(Ks, DDIM, ct * 16, ks * 32, lane);
      s0 = wmma_f16(a0, b0, s0);
      v16h a1 = load_frag_a(Qs, DDIM, mt * 16, (ks + 1) * 32, lane);
      v16h b1 = load_frag_b(Ks, DDIM, ct * 16, (ks + 1) * 32, lane);
      s1 = wmma_f16(a1, b1, s1);
    }
    #pragma unroll
    for (int v = 0; v < 8; ++v) {
      int row = mt * 16 + v + ((lane >> 4) << 3);
      int col = ct * 16 + (lane & 15);
      Sraw[row * KT + col] = s0[v] + s1[v];
    }
    __syncthreads();

    // ---- Online softmax: 8 threads per row, 8 cols each ----
    {
      int r   = tid >> 3;
      int sub = tid & 7;
      int c0  = sub * 8;
      float vals[8];
      float mv = -1e30f;
      #pragma unroll
      for (int j = 0; j < 8; ++j) {
        vals[j] = Sraw[r * KT + c0 + j];
        mv = fmaxf(mv, vals[j]);
      }
      #pragma unroll
      for (int off = 4; off > 0; off >>= 1)
        mv = fmaxf(mv, __shfl_xor(mv, off, 32));
      float mprev = mArr[r];
      float mnew  = fmaxf(mprev, mv);
      float sc    = __expf(mprev - mnew);
      float sum = 0.0f;
      #pragma unroll
      for (int j = 0; j < 8; ++j) {
        float p = __expf(vals[j] - mnew);
        Ps[r * KT + c0 + j] = (_Float16)p;
        sum += p;
      }
      #pragma unroll
      for (int off = 4; off > 0; off >>= 1)
        sum += __shfl_xor(sum, off, 32);
      if (sub == 0) {
        lArr[r]  = lArr[r] * sc + sum;
        mArr[r]  = mnew;
        scArr[r] = sc;
      }
    }
    __syncthreads();

    // ---- Rescale O accumulators by per-row scale ----
    #pragma unroll
    for (int mt2 = 0; mt2 < 2; ++mt2) {
      #pragma unroll
      for (int v = 0; v < 8; ++v) {
        float sc = scArr[mt2 * 16 + v + ((lane >> 4) << 3)];
        #pragma unroll
        for (int nt = 0; nt < 4; ++nt)
          accO[mt2 * 4 + nt][v] *= sc;
      }
    }

    // ---- O += P @ V  (k over KT=64, wave's 64-wide D slice) ----
    v16h afr[2][2];
    #pragma unroll
    for (int mt2 = 0; mt2 < 2; ++mt2)
      #pragma unroll
      for (int ks = 0; ks < 2; ++ks)
        afr[mt2][ks] = load_frag_a(Ps, KT, mt2 * 16, ks * 32, lane);
    #pragma unroll
    for (int ks = 0; ks < 2; ++ks) {
      #pragma unroll
      for (int nt = 0; nt < 4; ++nt) {
        v16h bfr = load_frag_b(Vt, KT, dbase + nt * 16, ks * 32, lane);
        #pragma unroll
        for (int mt2 = 0; mt2 < 2; ++mt2)
          accO[mt2 * 4 + nt] = wmma_f16(afr[mt2][ks], bfr, accO[mt2 * 4 + nt]);
      }
    }
  }

  // ---- Normalize and write output ----
  float* ob = out + ((size_t)b * S_LEN + q0) * DDIM;
  #pragma unroll
  for (int mt2 = 0; mt2 < 2; ++mt2) {
    float il[8];
    #pragma unroll
    for (int v = 0; v < 8; ++v)
      il[v] = 1.0f / lArr[mt2 * 16 + v + ((lane >> 4) << 3)];
    #pragma unroll
    for (int nt = 0; nt < 4; ++nt) {
      #pragma unroll
      for (int v = 0; v < 8; ++v) {
        int row = mt2 * 16 + v + ((lane >> 4) << 3);
        int col = dbase + nt * 16 + (lane & 15);
        ob[(size_t)row * DDIM + col] = accO[mt2 * 4 + nt][v] * il[v];
      }
    }
  }
}

extern "C" void kernel_launch(void* const* d_in, const int* in_sizes, int n_in,
                              void* d_out, int out_size, void* d_ws, size_t ws_size,
                              hipStream_t stream) {
  (void)in_sizes; (void)n_in; (void)out_size; (void)d_ws; (void)ws_size;
  const float* x = (const float*)d_in[0];
  float* out = (float*)d_out;

  hipFuncSetAttribute((const void*)attn_flash_kernel,
                      hipFuncAttributeMaxDynamicSharedMemorySize, SMEM_BYTES);

  dim3 grid(S_LEN / QT, 8);   // 64 q-tiles x 8 batches
  dim3 block(256);            // 8 wave32
  attn_flash_kernel<<<grid, block, SMEM_BYTES, stream>>>(x, out);
}
